// DGL_AttentiveFP_1692217114866
// MI455X (gfx1250) — matrix operand order repaired and checked by reading, not compiled, not run on hardware
//
#include <hip/hip_runtime.h>
#include <math.h>

// ---------------- types ----------------
typedef __attribute__((ext_vector_type(16))) __bf16          v16bf;
typedef __attribute__((ext_vector_type(8)))  float           v8f;
typedef __attribute__((ext_vector_type(16))) unsigned short  v16us;
typedef __attribute__((ext_vector_type(8)))  unsigned short  v8us;
typedef __attribute__((ext_vector_type(4)))  unsigned short  v4us;

#define ACT_NONE  0
#define ACT_LEAKY 1
#define ACT_RELU  2
#define ACT_ELU   3

template <int A>
__device__ __forceinline__ float actf_t(float x) {
  if (A == ACT_LEAKY) return x > 0.f ? x : 0.01f * x;
  if (A == ACT_RELU)  return fmaxf(x, 0.f);
  if (A == ACT_ELU)   return x > 0.f ? x : (__expf(x) - 1.f);
  return x;
}

__device__ __forceinline__ unsigned short f2bfu(float f) {  // f32 -> bf16 (RNE)
  unsigned u = __float_as_uint(f);
  unsigned r = u + 0x7FFFu + ((u >> 16) & 1u);
  return (unsigned short)(r >> 16);
}

// ---------------- LDS-staged WMMA GEMM:  C = OUT_ACT(IN_ACT(A) @ W + bias) ----------------
// A: [Ma,lda] f32 row-major (cols K..lda-1 finite if lda>K), W: [K,Nc] f32, C: [Ma,Nc] f32.
// Block: 256 threads (8 waves). Block tile 128(M) x 128(N), K-step 32.
// Waves 4(M) x 2(N); each wave owns a 32x64 macro-tile = 2x4 WMMA tiles
// (8 v_wmma per K-step, A-fragments amortized over 4 B-tiles).
// A staged bf16 [row][k] (stride 40us=80B keeps 16B alignment); B staged transposed [n][k].
// Fragment halves are 16 contiguous LDS bytes -> two ds_load_b128 per fragment.
// VEC4 path (lda%4==0, Nc%4==0): A stage = 4x b128 global + 4x b64 LDS stores,
// B stage = 4x b128 global (contiguous in N) + 16x b16 transposed LDS stores.
#define BM 128
#define BN 128
#define BK 32
#define ASTR 40
#define BSTR 40

template <int IN_ACT, int OUT_ACT, bool VEC4>
__global__ void __launch_bounds__(256)
k_wmma_gemm(const float* __restrict__ A, const float* __restrict__ Wt,
            const float* __restrict__ bias, float* __restrict__ C,
            int Ma, int K, int Nc, int lda) {
  __shared__ unsigned short Alds[BM * ASTR];   // 10240 B
  __shared__ unsigned short Blds[BN * BSTR];   // 10240 B

  const int tid  = threadIdx.x;
  const int lane = tid & 31;
  const int wave = tid >> 5;
  const int half = lane >> 4;
  const int l15  = lane & 15;
  const int wm   = (wave & 3) * 32;    // wave m-offset in block tile
  const int wn   = (wave >> 2) * 64;   // wave n-offset in block tile
  const int m0b  = blockIdx.y * BM;
  const int n0b  = blockIdx.x * BN;

  // staging assignments
  const int ar    = tid >> 1;          // A row 0..127
  const int ahalf = (tid & 1) * 16;    // A k-subrange (16 wide)
  const bool arok = (m0b + ar) < Ma;
  const float* arow = A + (size_t)(m0b + ar) * lda;
  // scalar-path B: thread covers col bn, 16 k's
  const int bns  = tid >> 1;           // B col 0..127
  const int bkq  = (tid & 1) * 16;
  const bool bnok = (n0b + bns) < Nc;
  // vec4-path B: thread covers k=bk, 16 consecutive n
  const int bk   = tid >> 3;           // k 0..31
  const int bnq  = (tid & 7) * 16;     // 16 consecutive n

  v8f acc[2][4] = {{{}, {}, {}, {}}, {{}, {}, {}, {}}};

  for (int k0 = 0; k0 < K; k0 += BK) {
    __builtin_prefetch(arow + k0 + BK, 0, 3);   // speculative-safe, near-cache hint
    if (VEC4) {
      // ---- A tile: 4x float4 loads, IN_ACT fused into f32->bf16 ----
#pragma unroll
      for (int g = 0; g < 4; ++g) {
        int k = k0 + ahalf + 4 * g;
        float4 v = make_float4(0.f, 0.f, 0.f, 0.f);
        if (arok && k < K) v = *(const float4*)(arow + k);
        v4us u;
        u[0] = f2bfu(actf_t<IN_ACT>(v.x)); u[1] = f2bfu(actf_t<IN_ACT>(v.y));
        u[2] = f2bfu(actf_t<IN_ACT>(v.z)); u[3] = f2bfu(actf_t<IN_ACT>(v.w));
        *(v4us*)&Alds[ar * ASTR + ahalf + 4 * g] = u;
      }
      // ---- B tile: contiguous-N loads, transposed b16 stores ----
      {
        int kk = k0 + bk;
        bool kok = kk < K;
        const float* wrow = Wt + (size_t)kk * Nc;
#pragma unroll
        for (int g = 0; g < 4; ++g) {
          int n = n0b + bnq + 4 * g;
          float4 v = make_float4(0.f, 0.f, 0.f, 0.f);
          if (kok && n < Nc) v = *(const float4*)(wrow + n);
          Blds[(bnq + 4 * g + 0) * BSTR + bk] = f2bfu(v.x);
          Blds[(bnq + 4 * g + 1) * BSTR + bk] = f2bfu(v.y);
          Blds[(bnq + 4 * g + 2) * BSTR + bk] = f2bfu(v.z);
          Blds[(bnq + 4 * g + 3) * BSTR + bk] = f2bfu(v.w);
        }
      }
    } else {
      // ---- scalar fallback (only the K=39 node-projection GEMM) ----
      bool afull = arok && (k0 + BK <= K);
      if (afull) {
#pragma unroll
        for (int i = 0; i < 16; ++i)
          Alds[ar * ASTR + ahalf + i] = f2bfu(actf_t<IN_ACT>(arow[k0 + ahalf + i]));
      } else {
#pragma unroll
        for (int i = 0; i < 16; ++i) {
          int k = k0 + ahalf + i;
          float v = (arok && k < K) ? actf_t<IN_ACT>(arow[k]) : 0.f;
          Alds[ar * ASTR + ahalf + i] = f2bfu(v);
        }
      }
#pragma unroll
      for (int i = 0; i < 16; ++i) {
        int k = k0 + bkq + i;
        float v = (bnok && k < K) ? Wt[(size_t)k * Nc + (n0b + bns)] : 0.f;
        Blds[bns * BSTR + bkq + i] = f2bfu(v);
      }
    }
    __syncthreads();

    // ---- fragments from LDS (two 16B ds reads each) ----
    v16bf afrag[2], bfrag[4];
#pragma unroll
    for (int i = 0; i < 2; ++i) {
      const unsigned short* p = &Alds[(wm + i * 16 + l15) * ASTR];
      v8us a0 = *(const v8us*)(p + 8 * half);        // k = 8*half + 0..7
      v8us a1 = *(const v8us*)(p + 16 + 8 * half);   // k = 16+8*half + 0..7
      afrag[i] = __builtin_bit_cast(v16bf, __builtin_shufflevector(
          a0, a1, 0,1,2,3,4,5,6,7,8,9,10,11,12,13,14,15));
    }
#pragma unroll
    for (int j = 0; j < 4; ++j) {
      const unsigned short* p = &Blds[(wn + j * 16 + l15) * BSTR];
      v8us b0 = *(const v8us*)(p + 8 * half);
      v8us b1 = *(const v8us*)(p + 16 + 8 * half);
      bfrag[j] = __builtin_bit_cast(v16bf, __builtin_shufflevector(
          b0, b1, 0,1,2,3,4,5,6,7,8,9,10,11,12,13,14,15));
    }
#pragma unroll
    for (int i = 0; i < 2; ++i)
#pragma unroll
      for (int j = 0; j < 4; ++j)
        acc[i][j] = __builtin_amdgcn_wmma_f32_16x16x32_bf16(
            false, afrag[i], false, bfrag[j], (short)0, acc[i][j], false, false);
    __syncthreads();
  }

  // ---- epilogue: lanes 0-15 -> N=lane,M=r ; lanes 16-31 -> N=lane-16,M=r+8 ----
#pragma unroll
  for (int i = 0; i < 2; ++i)
#pragma unroll
    for (int j = 0; j < 4; ++j)
#pragma unroll
      for (int r = 0; r < 8; ++r) {
        int m = m0b + wm + i * 16 + r + 8 * half;
        int n = n0b + wn + j * 16 + l15;
        if (m < Ma && n < Nc)
          C[(size_t)m * Nc + n] = actf_t<OUT_ACT>(acc[i][j][r] + bias[n]);
      }
}

// ---------------- elementwise / graph kernels ----------------
__global__ void k_fill(float* __restrict__ p, size_t n, float v) {
  size_t i = (size_t)blockIdx.x * blockDim.x + threadIdx.x;
  if (i < n) p[i] = v;
}

// X1[e] = concat(node[src[e]] (F), edge[e] (E), zero-pad to FEP)
__global__ void k_concat_gather(const float* __restrict__ node, const float* __restrict__ edgef,
                                const int* __restrict__ src, float* __restrict__ X1,
                                int M, int F, int E, int FEP) {
  size_t i = (size_t)blockIdx.x * blockDim.x + threadIdx.x;
  if (i >= (size_t)M * FEP) return;
  int e = (int)(i / FEP), j = (int)(i % FEP);
  float v = 0.f;
  if (j < F) v = node[(size_t)src[e] * F + j];
  else if (j < F + E) v = edgef[(size_t)e * E + (j - F)];
  X1[i] = v;
}

// out[i] = leaky( dot(ACTU(U[idxU[i]]), w[0:G]) + dot(V[idxV[i]], w[G:2G]) + b ), float4 inner loop
template <int ACTU>
__global__ void k_rowdot2(const float* __restrict__ U, const int* __restrict__ idxU,
                          const float* __restrict__ V, const int* __restrict__ idxV,
                          const float* __restrict__ w, const float* __restrict__ bptr,
                          float* __restrict__ out, int n, int G) {
  size_t i = (size_t)blockIdx.x * blockDim.x + threadIdx.x;
  if (i >= (size_t)n) return;
  int ru = idxU ? idxU[i] : (int)i;
  int rv = idxV ? idxV[i] : (int)i;
  const float4* u4 = (const float4*)(U + (size_t)ru * G);
  const float4* v4 = (const float4*)(V + (size_t)rv * G);
  const float4* wu = (const float4*)w;
  const float4* wv = (const float4*)(w + G);
  float s = bptr[0];
  const int G4 = G >> 2;
  for (int g = 0; g < G4; ++g) {
    float4 a = u4[g], b = v4[g], x = wu[g], y = wv[g];
    s += actf_t<ACTU>(a.x) * x.x + actf_t<ACTU>(a.y) * x.y +
         actf_t<ACTU>(a.z) * x.z + actf_t<ACTU>(a.w) * x.w;
    s += b.x * y.x + b.y * y.y + b.z * y.z + b.w * y.w;
  }
  out[i] = s > 0.f ? s : 0.01f * s;
}

// segment max via int-punned atomics (init smax to -inf first)
__global__ void k_seg_max(const float* __restrict__ v, const int* __restrict__ seg,
                          float* __restrict__ smax, int n) {
  size_t i = (size_t)blockIdx.x * blockDim.x + threadIdx.x;
  if (i >= (size_t)n) return;
  float x = v[i]; int s = seg[i];
  if (x >= 0.f) atomicMax((int*)&smax[s], __float_as_int(x));
  else          atomicMin((unsigned int*)&smax[s], __float_as_uint(x));
}

// e[i] = exp(v[i]-smax[seg]); ssum[seg] += e[i]  (init ssum=0 first)
__global__ void k_exp_sum(const float* __restrict__ v, const int* __restrict__ seg,
                          const float* __restrict__ smax, float* __restrict__ eexp,
                          float* __restrict__ ssum, int n) {
  size_t i = (size_t)blockIdx.x * blockDim.x + threadIdx.x;
  if (i >= (size_t)n) return;
  int s = seg[i];
  float e = __expf(v[i] - smax[s]);
  eexp[i] = e;
  atomicAdd(&ssum[s], e);
}

// out[seg[i]] += (e[i]/ssum[seg[i]]) * Vmat[rowidx?rowidx[i]:i]; quad-per-thread
__global__ void k_scatter(const float* __restrict__ Vmat, const int* __restrict__ rowidx,
                          const float* __restrict__ eexp, const float* __restrict__ ssum,
                          const int* __restrict__ seg, float* __restrict__ out,
                          int n, int G) {
  const int G4 = G >> 2;
  size_t t = (size_t)blockIdx.x * blockDim.x + threadIdx.x;
  if (t >= (size_t)n * G4) return;
  int i = (int)(t / G4), q = (int)(t % G4);
  int d = seg[i];
  int r = rowidx ? rowidx[i] : i;
  float a = eexp[i] / ssum[d];
  float4 v = *(const float4*)(Vmat + (size_t)r * G + 4 * q);
  float* o = out + (size_t)d * G + 4 * q;
  atomicAdd(o + 0, a * v.x); atomicAdd(o + 1, a * v.y);
  atomicAdd(o + 2, a * v.z); atomicAdd(o + 3, a * v.w);
}

// out[seg[i]] += H[i]; quad-per-thread
__global__ void k_segsum_rows(const float* __restrict__ H, const int* __restrict__ seg,
                              float* __restrict__ out, int n, int G) {
  const int G4 = G >> 2;
  size_t t = (size_t)blockIdx.x * blockDim.x + threadIdx.x;
  if (t >= (size_t)n * G4) return;
  int i = (int)(t / G4), q = (int)(t % G4);
  float4 v = *(const float4*)(H + (size_t)i * G + 4 * q);
  float* o = out + (size_t)seg[i] * G + 4 * q;
  atomicAdd(o + 0, v.x); atomicAdd(o + 1, v.y);
  atomicAdd(o + 2, v.z); atomicAdd(o + 3, v.w);
}

// GRUCell elementwise: gates precomputed gx=x@wih+bih, gh=h@whh+bhh (each [rows,3H])
__global__ void k_gru(const float* __restrict__ gx, const float* __restrict__ gh,
                      const float* __restrict__ h, float* __restrict__ out,
                      int rows, int H, int relu_out) {
  size_t t = (size_t)blockIdx.x * blockDim.x + threadIdx.x;
  if (t >= (size_t)rows * H) return;
  int i = (int)(t / H), j = (int)(t % H);
  size_t b = (size_t)i * 3 * H;
  float r = 1.f / (1.f + __expf(-(gx[b + j]     + gh[b + j])));
  float z = 1.f / (1.f + __expf(-(gx[b + H + j] + gh[b + H + j])));
  float nn = tanhf(gx[b + 2 * H + j] + r * gh[b + 2 * H + j]);
  float o = (1.f - z) * nn + z * h[t];
  if (relu_out) o = fmaxf(o, 0.f);
  out[t] = o;
}

// ---------------- host side ----------------
static inline unsigned cdiv(size_t a, size_t b) { return (unsigned)((a + b - 1) / b); }

// Only the (IN_ACT, OUT_ACT, VEC4) combos actually used are instantiated:
//   (NONE,LEAKY,false)  node projection (K=39)
//   (NONE,LEAKY,true)   edge projection (padded X1)
//   (NONE,NONE,true)    plain projections / gate GEMMs
//   (ELU,NONE,true)     GRU x-gate GEMMs with fused elu on input
static void gemm(const float* A, int lda, const float* W, const float* b, float* C,
                 int Ma, int K, int Nc, int in_act, int out_act, hipStream_t s) {
  dim3 blk(256, 1, 1), grd(cdiv(Nc, BN), cdiv(Ma, BM), 1);
  bool vec4 = (lda % 4 == 0) && (Nc % 4 == 0);
  if (in_act == ACT_ELU)
    k_wmma_gemm<ACT_ELU, ACT_NONE, true ><<<grd, blk, 0, s>>>(A, W, b, C, Ma, K, Nc, lda);
  else if (out_act == ACT_LEAKY && vec4)
    k_wmma_gemm<ACT_NONE, ACT_LEAKY, true ><<<grd, blk, 0, s>>>(A, W, b, C, Ma, K, Nc, lda);
  else if (out_act == ACT_LEAKY)
    k_wmma_gemm<ACT_NONE, ACT_LEAKY, false><<<grd, blk, 0, s>>>(A, W, b, C, Ma, K, Nc, lda);
  else
    k_wmma_gemm<ACT_NONE, ACT_NONE, true ><<<grd, blk, 0, s>>>(A, W, b, C, Ma, K, Nc, lda);
}
static void fillf(float* p, size_t n, float v, hipStream_t s) {
  k_fill<<<cdiv(n, 256), 256, 0, s>>>(p, n, v);
}

extern "C" void kernel_launch(void* const* d_in, const int* in_sizes, int n_in,
                              void* d_out, int out_size, void* d_ws, size_t ws_size,
                              hipStream_t stream) {
  (void)in_sizes; (void)n_in; (void)out_size; (void)ws_size;
  const int N = 50000, M = 400000, Bg = 1000;
  const int F = 39, E = 11, G = 200, P = 128;
  const int FEP = 52;  // F+E=50 zero-padded to multiple of 4 for the VEC4 GEMM path

  const float* node   = (const float*)d_in[0];
  const float* edgef  = (const float*)d_in[1];
  const float* pn_w   = (const float*)d_in[2];
  const float* pn_b   = (const float*)d_in[3];
  const float* pe1_w  = (const float*)d_in[4];
  const float* pe1_b  = (const float*)d_in[5];
  const float* pe2_w  = (const float*)d_in[6];
  const float* pe2_b  = (const float*)d_in[7];
  const float* et_w   = (const float*)d_in[8];
  const float* et_b   = (const float*)d_in[9];
  const float* g0_wih = (const float*)d_in[10];
  const float* g0_whh = (const float*)d_in[11];
  const float* g0_bih = (const float*)d_in[12];
  const float* g0_bhh = (const float*)d_in[13];
  const float* lpe_w  = (const float*)d_in[14];
  const float* lpe_b  = (const float*)d_in[15];
  const float* lpn_w  = (const float*)d_in[16];
  const float* lpn_b  = (const float*)d_in[17];
  const float* g1_wih = (const float*)d_in[18];
  const float* g1_whh = (const float*)d_in[19];
  const float* g1_bih = (const float*)d_in[20];
  const float* g1_bhh = (const float*)d_in[21];
  const float* rl_w   = (const float*)d_in[22];  // [T,2G]
  const float* rl_b   = (const float*)d_in[23];  // [T]
  const float* rp_w   = (const float*)d_in[24];  // [T,G,G]
  const float* rp_b   = (const float*)d_in[25];  // [T,G]
  const float* rg_wih = (const float*)d_in[26];  // [T,G,3G]
  const float* rg_whh = (const float*)d_in[27];
  const float* rg_bih = (const float*)d_in[28];  // [T,3G]
  const float* rg_bhh = (const float*)d_in[29];
  const float* t_w    = (const float*)d_in[30];
  const float* t_b    = (const float*)d_in[31];
  const int*   src    = (const int*)d_in[32];
  const int*   dst    = (const int*)d_in[33];
  const int*   gid    = (const int*)d_in[34];

  // workspace bump allocator (all sizes multiples of 4 floats -> 16B-aligned rows)
  float* Wp = (float*)d_ws;
  size_t o = 0;
  auto alloc = [&](size_t n) { float* p = Wp + o; o += n; return p; };
  float* hv   = alloc((size_t)N * G);       // hv_new; reused for hv_proj / readout hv
  float* h1   = alloc((size_t)N * G);
  float* h2   = alloc((size_t)N * G);
  float* cbuf = alloc((size_t)N * G);
  float* gx   = alloc((size_t)N * 3 * G);
  float* gh   = alloc((size_t)N * 3 * G);
  float* he1  = alloc((size_t)M * G);
  float* met  = alloc((size_t)M * G);
  float* X1   = alloc((size_t)M * FEP);
  float* elog = alloc((size_t)M);
  float* eexp = alloc((size_t)M);
  float* smax = alloc((size_t)N);
  float* ssum = alloc((size_t)N);
  float* gA   = alloc((size_t)Bg * G);
  float* gB   = alloc((size_t)Bg * G);
  float* grep = alloc((size_t)Bg * G);
  float* gxg  = alloc((size_t)Bg * 3 * G);
  float* ghg  = alloc((size_t)Bg * 3 * G);

  // ============ GetContext (AttentiveFP layer 0) ============
  gemm(node, F, pn_w, pn_b, hv, N, F, G, ACT_NONE, ACT_LEAKY, stream);       // scalar path (K=39)
  k_concat_gather<<<cdiv((size_t)M * FEP, 256), 256, 0, stream>>>(node, edgef, src, X1, M, F, E, FEP);
  gemm(X1, FEP, pe1_w, pe1_b, he1, M, F + E, G, ACT_NONE, ACT_LEAKY, stream);
  k_rowdot2<ACT_NONE><<<cdiv(M, 256), 256, 0, stream>>>(hv, dst, he1, nullptr, pe2_w, pe2_b, elog, M, G);
  fillf(smax, N, -INFINITY, stream);
  k_seg_max<<<cdiv(M, 256), 256, 0, stream>>>(elog, dst, smax, M);
  fillf(ssum, N, 0.f, stream);
  k_exp_sum<<<cdiv(M, 256), 256, 0, stream>>>(elog, dst, smax, eexp, ssum, M);
  gemm(he1, G, et_w, et_b, met, M, G, G, ACT_NONE, ACT_NONE, stream);        // biggest GEMM
  fillf(cbuf, (size_t)N * G, 0.f, stream);
  k_scatter<<<cdiv((size_t)M * (G / 4), 256), 256, 0, stream>>>(met, nullptr, eexp, ssum, dst, cbuf, M, G);
  gemm(cbuf, G, g0_wih, g0_bih, gx, N, G, 3 * G, ACT_ELU, ACT_NONE, stream);
  gemm(hv,   G, g0_whh, g0_bhh, gh, N, G, 3 * G, ACT_NONE, ACT_NONE, stream);
  k_gru<<<cdiv((size_t)N * G, 256), 256, 0, stream>>>(gx, gh, hv, h1, N, G, 1);

  // ============ GNNLayer ============
  k_rowdot2<ACT_NONE><<<cdiv(M, 256), 256, 0, stream>>>(h1, dst, h1, src, lpe_w, lpe_b, elog, M, G);
  fillf(smax, N, -INFINITY, stream);
  k_seg_max<<<cdiv(M, 256), 256, 0, stream>>>(elog, dst, smax, M);
  fillf(ssum, N, 0.f, stream);
  k_exp_sum<<<cdiv(M, 256), 256, 0, stream>>>(elog, dst, smax, eexp, ssum, M);
  gemm(h1, G, lpn_w, lpn_b, hv, N, G, G, ACT_NONE, ACT_NONE, stream);        // hv_proj (reuse hv)
  fillf(cbuf, (size_t)N * G, 0.f, stream);
  k_scatter<<<cdiv((size_t)M * (G / 4), 256), 256, 0, stream>>>(hv, src, eexp, ssum, dst, cbuf, M, G);
  gemm(cbuf, G, g1_wih, g1_bih, gx, N, G, 3 * G, ACT_ELU, ACT_NONE, stream);
  gemm(h1,   G, g1_whh, g1_bhh, gh, N, G, 3 * G, ACT_NONE, ACT_NONE, stream);
  k_gru<<<cdiv((size_t)N * G, 256), 256, 0, stream>>>(gx, gh, h1, h2, N, G, 1);

  // ============ AttentiveFPReadout (T=2) ============
  fillf(gA, (size_t)Bg * G, 0.f, stream);
  k_segsum_rows<<<cdiv((size_t)N * (G / 4), 256), 256, 0, stream>>>(h2, gid, gA, N, G);
  float* gcur = gA; float* gnext = gB;
  for (int t = 0; t < 2; ++t) {
    k_rowdot2<ACT_RELU><<<cdiv(N, 256), 256, 0, stream>>>(gcur, gid, h2, nullptr,
                                                          rl_w + (size_t)t * 2 * G, rl_b + t, elog, N, G);
    fillf(smax, Bg, -INFINITY, stream);
    k_seg_max<<<cdiv(N, 256), 256, 0, stream>>>(elog, gid, smax, N);
    fillf(ssum, Bg, 0.f, stream);
    k_exp_sum<<<cdiv(N, 256), 256, 0, stream>>>(elog, gid, smax, eexp, ssum, N);
    gemm(h2, G, rp_w + (size_t)t * G * G, rp_b + (size_t)t * G, hv, N, G, G, ACT_NONE, ACT_NONE, stream);
    fillf(grep, (size_t)Bg * G, 0.f, stream);
    k_scatter<<<cdiv((size_t)N * (G / 4), 256), 256, 0, stream>>>(hv, nullptr, eexp, ssum, gid, grep, N, G);
    gemm(grep, G, rg_wih + (size_t)t * G * 3 * G, rg_bih + (size_t)t * 3 * G, gxg, Bg, G, 3 * G, ACT_ELU, ACT_NONE, stream);
    gemm(gcur, G, rg_whh + (size_t)t * G * 3 * G, rg_bhh + (size_t)t * 3 * G, ghg, Bg, G, 3 * G, ACT_NONE, ACT_NONE, stream);
    k_gru<<<cdiv((size_t)Bg * G, 256), 256, 0, stream>>>(gxg, ghg, gcur, gnext, Bg, G, 0);
    float* tmp = gcur; gcur = gnext; gnext = tmp;
  }

  // ============ final transform ============
  gemm(gcur, G, t_w, t_b, (float*)d_out, Bg, G, P, ACT_NONE, ACT_NONE, stream);
}
